// MTVAE_10050223473246
// MI455X (gfx1250) — compile-verified, matrix operand order checked by reading, not run on hardware
//
#include <hip/hip_runtime.h>
#include <hip/hip_bf16.h>

#define Bv 256
#define Nv 50
#define Fv 6144
#define Hv 4
#define Ov 64
#define ALPHA 0.2f
#define BETA 0.3f

#define TM 128          // GEMM block tile rows
#define LDK 40          // padded LDS row stride (bf16 elems), keeps 16-B alignment

typedef __attribute__((ext_vector_type(2)))  __bf16 v2bf;
typedef __attribute__((ext_vector_type(4)))  __bf16 v4bf;
typedef __attribute__((ext_vector_type(8)))  __bf16 v8bf;
typedef __attribute__((ext_vector_type(16))) __bf16 v16bf;
typedef __attribute__((ext_vector_type(8)))  float  v8f;

// Convert+pack two f32 -> one VGPR of 2x bf16.
// Preferred: packed-cvt builtin (if toolchain declares it). Otherwise use the
// CDNA5 VOP3P V_FMA_MIXLO_BF16 / V_FMA_MIXHI_BF16 pair (ISA 15.10 op 62/63):
// with src1=1.0, src2=0 and default op_sel (f32 inputs), this is a pure
// RNE f32->bf16 conversion into the low/high half of the destination.
__device__ __forceinline__ v2bf pack_bf16(float lo, float hi) {
#if __has_builtin(__builtin_amdgcn_cvt_pk_bf16_f32)
    auto r = __builtin_amdgcn_cvt_pk_bf16_f32(lo, hi);
    v2bf o;
    __builtin_memcpy(&o, &r, 4);
    return o;
#else
    unsigned pk;
    asm("v_fma_mixlo_bf16 %0, %1, 1.0, 0\n\t"
        "v_fma_mixhi_bf16 %0, %2, 1.0, 0"
        : "=&v"(pk)
        : "v"(lo), "v"(hi));
    v2bf o;
    __builtin_memcpy(&o, &pk, 4);
    return o;
#endif
}

// ---------------------------------------------------------------------------
// Kernel 1: Gumbel-softmax hard adjacency.  Forward value of `edges` is
// exactly y_hard, and argmax is invariant under /TAU, so
// adj = (l0+g0 >= l1+g1) ? 1 : 0.
// ---------------------------------------------------------------------------
__global__ void adj_kernel(const float* __restrict__ probs,
                           const float* __restrict__ u,
                           float* __restrict__ adj, int total) {
    int i = blockIdx.x * blockDim.x + threadIdx.x;
    if (i >= total) return;
    float p = probs[i];
    // softmax([p, 1-p])[0] = sigmoid(2p-1)
    float prob0 = 1.f / (1.f + __expf(1.f - 2.f * p));
    float l0 = __logf(prob0 + 1e-5f);
    float l1 = __logf((1.f - prob0) + 1e-5f);
    float u0 = u[2 * i];
    float u1 = u[2 * i + 1];
    float g0 = -__logf(-__logf(u0 + 1e-10f) + 1e-10f);
    float g1 = -__logf(-__logf(u1 + 1e-10f) + 1e-10f);
    adj[i] = ((l0 + g0) >= (l1 + g1)) ? 1.f : 0.f;
}

// ---------------------------------------------------------------------------
// Kernel 2: Wh GEMM  (12800 x 6144) @ (6144 x 256), bf16 WMMA, f32 acc.
// Wh stored row-major [12800][256]: row = b*N+n, col = h*64+o (no div in
// epilogue). Grid (100, 4), block 256 = 8 waves. Block tile 128x64; wave
// tile 32x32 (4 accumulators, 4 WMMAs per K-step). Double-buffered LDS with
// one barrier per K-step; next tile is prefetched into registers while the
// current tile's WMMAs issue. f32->bf16 via v_fma_mixlo/hi_bf16.
// ---------------------------------------------------------------------------
__global__ __launch_bounds__(256)
void wh_gemm_kernel(const float* __restrict__ h,
                    const float* __restrict__ W,
                    float* __restrict__ Wh) {
    __shared__ __align__(16) __bf16 As[2][TM][LDK];  // [m][k]
    __shared__ __align__(16) __bf16 Bs[2][Ov][LDK];  // transposed: [o][k]

    const int tid   = threadIdx.x;
    const int lane  = tid & 31;
    const int w     = tid >> 5;
    const int waveM = w & 3;    // 4 row groups x 32 rows
    const int waveN = w >> 2;   // 2 col groups x 32 cols
    const int hh    = blockIdx.y;            // 64-col block == head
    const int rowBase = blockIdx.x * TM;

    v8f acc00 = {}, acc01 = {}, acc10 = {}, acc11 = {};

    // CDNA5 16-bit fragment addressing (ISA 7.12.2):
    // lanes 0-15 hold K {0..7, 16..23}; lanes 16-31 hold K {8..15, 24..31}
    const int kb0 = (lane >> 4) << 3;             // 0 or 8
    const int m0  = (waveM << 5) + (lane & 15);
    const int m1  = m0 + 16;
    const int n0  = (waveN << 5) + (lane & 15);
    const int n1  = n0 + 16;

    const int oB  = tid & 63;          // B-load: output column
    const int kbB = (tid >> 6) << 3;   // B-load: k-base (0,8,16,24)

    const float* hBase = h + (size_t)rowBase * Fv;
    const float* wBase = W + (size_t)hh * Fv * Ov;

    // ---- preload K-block 0 into buffer 0
    {
        int i = tid;
        #pragma unroll
        for (int it = 0; it < 4; ++it, i += 256) {
            int row = i >> 3, c4 = (i & 7) << 2;
            float4 v = *(const float4*)(hBase + (size_t)row * Fv + c4);
            union { v2bf h2[2]; v4bf v4; } t;
            t.h2[0] = pack_bf16(v.x, v.y);
            t.h2[1] = pack_bf16(v.z, v.w);
            *(v4bf*)&As[0][row][c4] = t.v4;
        }
        union { v2bf h2[4]; v8bf v8; } t;
        #pragma unroll
        for (int j = 0; j < 4; ++j)
            t.h2[j] = pack_bf16(wBase[(kbB + 2 * j) * Ov + oB],
                                wBase[(kbB + 2 * j + 1) * Ov + oB]);
        *(v8bf*)&Bs[0][oB][kbB] = t.v8;
    }
    __syncthreads();

    for (int kk = 0; kk < Fv; kk += 32) {
        const int  buf     = (kk >> 5) & 1;
        const bool hasNext = (kk + 32) < Fv;

        // ---- prefetch next K-block into registers (hides global latency)
        float4 pa[4];
        float  pb[8];
        if (hasNext) {
            int i = tid;
            #pragma unroll
            for (int it = 0; it < 4; ++it, i += 256) {
                int row = i >> 3, c4 = (i & 7) << 2;
                pa[it] = *(const float4*)(hBase + (size_t)row * Fv + (kk + 32) + c4);
            }
            const float* Wp = wBase + (size_t)(kk + 32) * Ov;
            #pragma unroll
            for (int j = 0; j < 8; ++j) pb[j] = Wp[(kbB + j) * Ov + oB];
        }

        // ---- compute on current buffer: 4 WMMAs (32x32 wave tile)
        union U { v8bf h2[2]; v16bf v; };
        U a0, a1, b0, b1;
        a0.h2[0] = *(const v8bf*)&As[buf][m0][kb0];
        a0.h2[1] = *(const v8bf*)&As[buf][m0][kb0 + 16];
        a1.h2[0] = *(const v8bf*)&As[buf][m1][kb0];
        a1.h2[1] = *(const v8bf*)&As[buf][m1][kb0 + 16];
        b0.h2[0] = *(const v8bf*)&Bs[buf][n0][kb0];
        b0.h2[1] = *(const v8bf*)&Bs[buf][n0][kb0 + 16];
        b1.h2[0] = *(const v8bf*)&Bs[buf][n1][kb0];
        b1.h2[1] = *(const v8bf*)&Bs[buf][n1][kb0 + 16];

        acc00 = __builtin_amdgcn_wmma_f32_16x16x32_bf16(
            false, a0.v, false, b0.v, (short)0, acc00, false, false);
        acc01 = __builtin_amdgcn_wmma_f32_16x16x32_bf16(
            false, a0.v, false, b1.v, (short)0, acc01, false, false);
        acc10 = __builtin_amdgcn_wmma_f32_16x16x32_bf16(
            false, a1.v, false, b0.v, (short)0, acc10, false, false);
        acc11 = __builtin_amdgcn_wmma_f32_16x16x32_bf16(
            false, a1.v, false, b1.v, (short)0, acc11, false, false);

        // ---- store prefetched tile into the other buffer
        if (hasNext) {
            const int nb = buf ^ 1;
            int i = tid;
            #pragma unroll
            for (int it = 0; it < 4; ++it, i += 256) {
                int row = i >> 3, c4 = (i & 7) << 2;
                union { v2bf h2[2]; v4bf v4; } t;
                t.h2[0] = pack_bf16(pa[it].x, pa[it].y);
                t.h2[1] = pack_bf16(pa[it].z, pa[it].w);
                *(v4bf*)&As[nb][row][c4] = t.v4;
            }
            union { v2bf h2[4]; v8bf v8; } t;
            #pragma unroll
            for (int j = 0; j < 4; ++j)
                t.h2[j] = pack_bf16(pb[2 * j], pb[2 * j + 1]);
            *(v8bf*)&Bs[nb][oB][kbB] = t.v8;
        }
        __syncthreads();
    }

    // ---- epilogue: Wh row-major [12800][256]; pure linear addressing
    const int colBase = (hh << 6) + (waveN << 5) + (lane & 15);
    #pragma unroll
    for (int r = 0; r < 8; ++r) {
        int mr = r + ((lane >> 4) << 3);       // C layout: lanes>=16 -> M+8
        size_t row0 = (size_t)(rowBase + (waveM << 5) + mr) * 256;
        size_t row1 = (size_t)(rowBase + (waveM << 5) + 16 + mr) * 256;
        Wh[row0 + colBase]      = acc00[r];
        Wh[row0 + colBase + 16] = acc01[r];
        Wh[row1 + colBase]      = acc10[r];
        Wh[row1 + colBase + 16] = acc11[r];
    }
}

// ---------------------------------------------------------------------------
// Kernel 3: per-(b,h) GAT attention. One workgroup per (b,h), 8 waves.
// Wh is row-major [12800][256]; the (b,h) tile is rows b*50.., cols h*64..
// ---------------------------------------------------------------------------
__global__ __launch_bounds__(256)
void attn_kernel(const float* __restrict__ Wh,
                 const float* __restrict__ a,
                 const float* __restrict__ adj,
                 float* __restrict__ out) {
    const int bh = blockIdx.x;
    const int b  = bh >> 2;
    const int hh = bh & 3;
    const int tid  = threadIdx.x;
    const int lane = tid & 31;
    const int w    = tid >> 5;

    __shared__ float Wh_s[Nv * Ov];
    __shared__ float a1_s[Nv * Ov];
    __shared__ float a2_s[Nv * Ov];
    __shared__ float e_s[Nv * 52];
    __shared__ float s1_s[Nv];

    const float* Whp = Wh + (size_t)b * Nv * 256 + (hh << 6);
    const float* ap  = a + (size_t)hh * Nv * (2 * Ov);
    for (int i = tid; i < Nv * Ov; i += 256) {
        int n = i >> 6, o = i & 63;
        Wh_s[i] = Whp[(size_t)n * 256 + o];
        a1_s[i] = ap[n * 128 + o];
        a2_s[i] = ap[n * 128 + 64 + o];
    }
    __syncthreads();

    // s1[n] = dot(Wh[n,:], a1[n,:])  -- one wave per row, shuffle reduce
    for (int n = w; n < Nv; n += 8) {
        float v = Wh_s[n * 64 + lane]      * a1_s[n * 64 + lane]
                + Wh_s[n * 64 + 32 + lane] * a1_s[n * 64 + 32 + lane];
        #pragma unroll
        for (int off = 16; off >= 1; off >>= 1) v += __shfl_xor(v, off, 32);
        if (lane == 0) s1_s[n] = v;
    }
    __syncthreads();

    // e[n,m] = mask(leaky(s1[n] + dot(a2[n,:], Wh[m,:])))
    const float* adjp = adj + (size_t)b * Nv * Nv;
    for (int idx = tid; idx < Nv * Nv; idx += 256) {
        int n = idx / Nv, m = idx % Nv;
        const float* a2r = &a2_s[n * 64];
        const float* whr = &Wh_s[m * 64];
        float acc = 0.f;
        #pragma unroll 8
        for (int o = 0; o < 64; ++o) acc = fmaf(a2r[o], whr[o], acc);
        float v = s1_s[n] + acc;
        v = (v >= 0.f) ? v : ALPHA * v;
        if (!(adjp[idx] > 0.f)) v = -9.0e15f;
        e_s[n * 52 + m] = v;
    }
    __syncthreads();

    // row softmax over m
    for (int n = w; n < Nv; n += 8) {
        float x1 = (lane < Nv)      ? e_s[n * 52 + lane]      : -3.4e38f;
        float x2 = (lane + 32 < Nv) ? e_s[n * 52 + lane + 32] : -3.4e38f;
        float mx = fmaxf(x1, x2);
        #pragma unroll
        for (int off = 16; off >= 1; off >>= 1) mx = fmaxf(mx, __shfl_xor(mx, off, 32));
        float ex1 = (lane < Nv)      ? __expf(x1 - mx) : 0.f;
        float ex2 = (lane + 32 < Nv) ? __expf(x2 - mx) : 0.f;
        float s = ex1 + ex2;
        #pragma unroll
        for (int off = 16; off >= 1; off >>= 1) s += __shfl_xor(s, off, 32);
        float inv = 1.f / s;
        if (lane < Nv)      e_s[n * 52 + lane]      = ex1 * inv;
        if (lane + 32 < Nv) e_s[n * 52 + lane + 32] = ex2 * inv;
    }
    __syncthreads();

    // h' = att @ Wh ; out = elu(h' + beta*Wh), stored as (B, N, H*O)
    for (int idx = tid; idx < Nv * Ov; idx += 256) {
        int n = idx >> 6, o = idx & 63;
        float acc = 0.f;
        #pragma unroll 5
        for (int m = 0; m < Nv; ++m) acc = fmaf(e_s[n * 52 + m], Wh_s[m * 64 + o], acc);
        float x = acc + BETA * Wh_s[idx];
        float r = (x > 0.f) ? x : (__expf(x) - 1.f);
        out[(((size_t)b * Nv + n) * Hv + hh) * Ov + o] = r;
    }
}

extern "C" void kernel_launch(void* const* d_in, const int* in_sizes, int n_in,
                              void* d_out, int out_size, void* d_ws, size_t ws_size,
                              hipStream_t stream) {
    const float* h     = (const float*)d_in[0];   // (B, N, F)
    const float* probs = (const float*)d_in[1];   // (B, N*N)
    const float* u     = (const float*)d_in[2];   // (B, N*N, 2)
    const float* W     = (const float*)d_in[3];   // (H, F, O)
    const float* a     = (const float*)d_in[4];   // (H, N, 2*O)
    float* out = (float*)d_out;

    float* adj = (float*)d_ws;                       // B*N*N floats
    float* Wh  = adj + (size_t)Bv * Nv * Nv;         // 12800*256 floats, row-major

    const int total = Bv * Nv * Nv;
    adj_kernel<<<(total + 255) / 256, 256, 0, stream>>>(probs, u, adj, total);
    wh_gemm_kernel<<<dim3((Bv * Nv) / TM, Hv), 256, 0, stream>>>(h, W, Wh);
    attn_kernel<<<Bv * Hv, 256, 0, stream>>>(Wh, a, adj, out);
}
